// MultiLatentAttention_64312840290360
// MI455X (gfx1250) — compile-verified, matrix operand order checked by reading, not run on hardware
//
#include <hip/hip_runtime.h>

typedef __bf16 bf16;
typedef __attribute__((ext_vector_type(16))) __bf16 v16bf;
typedef __attribute__((ext_vector_type(8)))  __bf16 v8bf;
typedef __attribute__((ext_vector_type(8)))  float  v8f;

#define D_      2048
#define B_      8
#define S_      4096
#define H_      16
#define HD_     128
#define L_      16
#define N_      256            // H_*L_ folded score columns
#define BS_     (B_*S_)        // 32768 rows
#define UCH_    32             // S-chunks for the p-bar weighted sum

union V16 { v16bf v; v8bf h[2]; };

// ---------------------------------------------------------------------------
// 1) LayerNorm -> bf16 h.  One 256-thread block per row; b128 loads/stores.
//    Thread t owns elements [t*8, t*8+8).
// ---------------------------------------------------------------------------
__global__ void ln_kernel(const float* __restrict__ x, const float* __restrict__ g,
                          const float* __restrict__ bta, bf16* __restrict__ h) {
    __shared__ float red[256];
    size_t row = blockIdx.x;
    int t = threadIdx.x;
    const float4* xr = (const float4*)(x + row * D_) + t * 2;
    float4 v0 = xr[0], v1 = xr[1];
    float s  = v0.x + v0.y + v0.z + v0.w + v1.x + v1.y + v1.z + v1.w;
    float s2 = v0.x*v0.x + v0.y*v0.y + v0.z*v0.z + v0.w*v0.w
             + v1.x*v1.x + v1.y*v1.y + v1.z*v1.z + v1.w*v1.w;
    red[t] = s; __syncthreads();
    for (int off = 128; off > 0; off >>= 1) { if (t < off) red[t] += red[t+off]; __syncthreads(); }
    float mean = red[0] * (1.0f / D_); __syncthreads();
    red[t] = s2; __syncthreads();
    for (int off = 128; off > 0; off >>= 1) { if (t < off) red[t] += red[t+off]; __syncthreads(); }
    float var  = red[0] * (1.0f / D_) - mean*mean;
    float rstd = rsqrtf(var + 1e-5f);

    const float4* gp = (const float4*)(g   + (size_t)t * 8);
    const float4* bp = (const float4*)(bta + (size_t)t * 8);
    float4 g0 = gp[0], g1 = gp[1], b0 = bp[0], b1 = bp[1];
    float in[8]  = { v0.x, v0.y, v0.z, v0.w, v1.x, v1.y, v1.z, v1.w };
    float gg[8]  = { g0.x, g0.y, g0.z, g0.w, g1.x, g1.y, g1.z, g1.w };
    float bb[8]  = { b0.x, b0.y, b0.z, b0.w, b1.x, b1.y, b1.z, b1.w };
    v8bf o;
    for (int i = 0; i < 8; ++i) o[i] = (bf16)((in[i] - mean) * rstd * gg[i] + bb[i]);
    *(v8bf*)(h + row * D_ + (size_t)t * 8) = o;
}

// ---------------------------------------------------------------------------
// 2) q = latents @ w_lq + b_lq    [16, 2048] f32
// ---------------------------------------------------------------------------
__global__ void q_kernel(const float* __restrict__ lat, const float* __restrict__ w,
                         const float* __restrict__ bias, float* __restrict__ q) {
    int j = blockIdx.x * 256 + threadIdx.x;   // grid.x = 8
    int l = blockIdx.y;                       // grid.y = 16
    float acc = bias[j];
    const float* lr = lat + l * D_;
    for (int d = 0; d < D_; ++d) acc += lr[d] * w[(size_t)d * D_ + j];
    q[l * D_ + j] = acc;
}

// ---------------------------------------------------------------------------
// 3) A-fold: At[n = h*16+l][k] = sum_d w_k[k, h*128+d] * q[l, h*128+d]  (bf16, [N,K])
// ---------------------------------------------------------------------------
__global__ void afold_kernel(const float* __restrict__ q, const float* __restrict__ w_k,
                             bf16* __restrict__ At) {
    int k = blockIdx.x * 256 + threadIdx.x;   // grid.x = 8
    int n = blockIdx.y;                       // grid.y = 256
    int hh = n >> 4, l = n & 15;
    const float* qr = q   + l * D_ + hh * HD_;
    const float* wr = w_k + (size_t)k * D_ + hh * HD_;
    float acc = 0.f;
    for (int d = 0; d < HD_; ++d) acc += qr[d] * wr[d];
    At[(size_t)n * D_ + k] = (bf16)acc;
}

// 3b) score bias constant: c[n] = q[l, hslice] . b_k[hslice]
__global__ void cbias_kernel(const float* __restrict__ q, const float* __restrict__ b_k,
                             float* __restrict__ c) {
    int n = threadIdx.x;                      // one block of 256
    int hh = n >> 4, l = n & 15;
    float acc = 0.f;
    for (int d = 0; d < HD_; ++d) acc += q[l * D_ + hh * HD_ + d] * b_k[hh * HD_ + d];
    c[n] = acc;
}

// ---------------------------------------------------------------------------
// 4) scores[BS, 256] = (h @ At^T + c) / sqrt(128), via WMMA bf16.
//    One 128-thread block (4 waves) per 16-row M-tile; each wave computes 4
//    adjacent N-tiles so the A operand is loaded once per K-step and reused
//    across 4 v_wmma.  h is streamed from HBM exactly once (128 MB).
//    Operand packing per CDNA5 ISA 7.12.2: lane m = lane&15, half = lane>>4;
//    vec elems 0..7 = K 8*half..+7, elems 8..15 = K 16+8*half..+7.
// ---------------------------------------------------------------------------
__global__ void scores_wmma_kernel(const bf16* __restrict__ h, const bf16* __restrict__ At,
                                   const float* __restrict__ c, float* __restrict__ scores) {
    int wave = threadIdx.x >> 5;              // 0..3 -> N columns [wave*64, wave*64+64)
    int lane = threadIdx.x & 31;
    int m0 = blockIdx.x * 16;                 // grid.x = 2048 M-tiles
    int mr = lane & 15, half = lane >> 4;

    const bf16* arow = h  + (size_t)(m0 + mr) * D_ + 8 * half;
    const bf16* brow = At + (size_t)(wave * 64 + mr) * D_ + 8 * half;

    v8f acc[4] = {};
    for (int k = 0; k < D_; k += 32) {
        V16 a;
        a.h[0] = *(const v8bf*)(arow + k);
        a.h[1] = *(const v8bf*)(arow + k + 16);
        V16 b[4];
        for (int i = 0; i < 4; ++i) {
            const bf16* br = brow + (size_t)(16 * i) * D_ + k;
            b[i].h[0] = *(const v8bf*)(br);
            b[i].h[1] = *(const v8bf*)(br + 16);
        }
        for (int i = 0; i < 4; ++i)
            acc[i] = __builtin_amdgcn_wmma_f32_16x16x32_bf16(
                         false, a.v, false, b[i].v, (short)0, acc[i], false, false);
    }
    const float scale = 0.08838834764831845f;     // 1/sqrt(128)
    int n = lane & 15;
    for (int i = 0; i < 4; ++i) {
        int n0 = wave * 64 + i * 16;
        float cn = c[n0 + n];
        for (int r = 0; r < 8; ++r) {
            int m = r + 8 * half;                 // D layout: lane n, VGPR r -> M = r + 8*(lane/16)
            scores[(size_t)(m0 + m) * N_ + (n0 + n)] = (acc[i][r] + cn) * scale;
        }
    }
}

// ---------------------------------------------------------------------------
// 5) softmax over s per (b,h,l), then pbar[b][s][h] = mean_l probs  (f32)
//    One block per (b,h); scores stay L2-resident (32MB < 192MB).
// ---------------------------------------------------------------------------
__global__ void softmax_kernel(const float* __restrict__ scores, float* __restrict__ pbar) {
    int b = blockIdx.x, hh = blockIdx.y;
    __shared__ float red[256 * 16];
    int t = threadIdx.x;
    const float* base = scores + (size_t)b * S_ * N_ + hh * 16;

    float mx[16];
    for (int l = 0; l < 16; ++l) mx[l] = -3.0e38f;
    for (int s = t; s < S_; s += 256) {
        const float4* p = (const float4*)(base + (size_t)s * N_);
        for (int i = 0; i < 4; ++i) {
            float4 v = p[i];
            mx[4*i+0] = fmaxf(mx[4*i+0], v.x); mx[4*i+1] = fmaxf(mx[4*i+1], v.y);
            mx[4*i+2] = fmaxf(mx[4*i+2], v.z); mx[4*i+3] = fmaxf(mx[4*i+3], v.w);
        }
    }
    for (int l = 0; l < 16; ++l) red[t*16+l] = mx[l];
    __syncthreads();
    for (int off = 128; off > 0; off >>= 1) {
        if (t < off) for (int l = 0; l < 16; ++l)
            red[t*16+l] = fmaxf(red[t*16+l], red[(t+off)*16+l]);
        __syncthreads();
    }
    float gmx[16]; for (int l = 0; l < 16; ++l) gmx[l] = red[l];
    __syncthreads();

    float sm[16]; for (int l = 0; l < 16; ++l) sm[l] = 0.f;
    for (int s = t; s < S_; s += 256) {
        const float4* p = (const float4*)(base + (size_t)s * N_);
        for (int i = 0; i < 4; ++i) {
            float4 v = p[i];
            sm[4*i+0] += __expf(v.x - gmx[4*i+0]); sm[4*i+1] += __expf(v.y - gmx[4*i+1]);
            sm[4*i+2] += __expf(v.z - gmx[4*i+2]); sm[4*i+3] += __expf(v.w - gmx[4*i+3]);
        }
    }
    for (int l = 0; l < 16; ++l) red[t*16+l] = sm[l];
    __syncthreads();
    for (int off = 128; off > 0; off >>= 1) {
        if (t < off) for (int l = 0; l < 16; ++l) red[t*16+l] += red[(t+off)*16+l];
        __syncthreads();
    }
    float inv[16]; for (int l = 0; l < 16; ++l) inv[l] = 1.0f / red[l];

    for (int s = t; s < S_; s += 256) {
        const float4* p = (const float4*)(base + (size_t)s * N_);
        float acc = 0.f;
        for (int i = 0; i < 4; ++i) {
            float4 v = p[i];
            acc += __expf(v.x - gmx[4*i+0]) * inv[4*i+0]
                 + __expf(v.y - gmx[4*i+1]) * inv[4*i+1]
                 + __expf(v.z - gmx[4*i+2]) * inv[4*i+2]
                 + __expf(v.w - gmx[4*i+3]) * inv[4*i+3];
        }
        pbar[((size_t)b * S_ + s) * H_ + hh] = acc * (1.0f / 16.0f);
    }
}

// ---------------------------------------------------------------------------
// 6) u[b,h,:] = sum_s pbar[b,h,s] * h[b,s,:]  -- split over S into UCH_ chunks
//    (deterministic two-stage reduction, no float atomics).
// ---------------------------------------------------------------------------
__global__ void upart_kernel(const bf16* __restrict__ h, const float* __restrict__ pbar,
                             float* __restrict__ upart) {
    int b = blockIdx.x, ch = blockIdx.y;      // grid (8, 32)
    int t = threadIdx.x;
    int s0 = ch * (S_ / UCH_);                // 128 s per chunk
    __shared__ float pl[128 * 16];
    for (int i = t; i < 128 * 16; i += 256)
        pl[i] = pbar[((size_t)b * S_ + s0) * H_ + i];
    __syncthreads();

    float acc[16][8];
    for (int hh = 0; hh < 16; ++hh) for (int j = 0; j < 8; ++j) acc[hh][j] = 0.f;
    for (int s = 0; s < 128; ++s) {
        const bf16* hr = h + ((size_t)b * S_ + s0 + s) * D_;
        float hv[8];
        for (int j = 0; j < 8; ++j) hv[j] = (float)hr[t + 256*j];
        for (int hh = 0; hh < 16; ++hh) {
            float p = pl[s * 16 + hh];
            for (int j = 0; j < 8; ++j) acc[hh][j] += p * hv[j];
        }
    }
    float* out = upart + (((size_t)b * UCH_ + ch) * H_) * D_;
    for (int hh = 0; hh < 16; ++hh)
        for (int j = 0; j < 8; ++j)
            out[(size_t)hh * D_ + t + 256*j] = acc[hh][j];
}

__global__ void ured_kernel(const float* __restrict__ upart, float* __restrict__ u) {
    int i = blockIdx.x * 256 + threadIdx.x;   // grid.x = 128 over H_*D_
    int b = blockIdx.y;                       // grid.y = 8
    float acc = 0.f;
    for (int ch = 0; ch < UCH_; ++ch)
        acc += upart[(((size_t)b * UCH_ + ch) * H_) * D_ + i];
    u[(size_t)b * H_ * D_ + i] = acc;
}

// ---------------------------------------------------------------------------
// 7) pc[b,j] = u[b, j>>7, :] . w_v[:, j] + b_v[j]   (sum_s pbar == 1 makes b_v exact)
// ---------------------------------------------------------------------------
__global__ void pc_kernel(const float* __restrict__ u, const float* __restrict__ w_v,
                          const float* __restrict__ b_v, float* __restrict__ pc) {
    int j = blockIdx.x * 256 + threadIdx.x;   // grid.x = 8
    int b = blockIdx.y;                       // grid.y = 8
    int hh = j >> 7;
    const float* ur = u + ((size_t)b * H_ + hh) * D_;
    float acc = b_v[j];
    for (int d = 0; d < D_; ++d) acc += ur[d] * w_v[(size_t)d * D_ + j];
    pc[b * D_ + j] = acc;
}

// 8) generic y[b,:] = x[b,:] @ W + bias  (used for w_lv then w_out)
__global__ void mv_kernel(const float* __restrict__ xin, const float* __restrict__ w,
                          const float* __restrict__ bias, float* __restrict__ y) {
    int j = blockIdx.x * 256 + threadIdx.x;   // grid.x = 8
    int b = blockIdx.y;                       // grid.y = 8
    const float* xr = xin + b * D_;
    float acc = bias[j];
    for (int d = 0; d < D_; ++d) acc += xr[d] * w[(size_t)d * D_ + j];
    y[b * D_ + j] = acc;
}

// 9) y = x + broadcast(outv[b])  (float4 vectorized)
__global__ void resid_kernel(const float* __restrict__ x, const float* __restrict__ outv,
                             float* __restrict__ y) {
    size_t idx = ((size_t)blockIdx.x * 256 + threadIdx.x) * 4;
    size_t col = idx & (D_ - 1);
    size_t b   = idx >> 23;                   // S_*D_ = 2^23 per batch
    float4 xv = *(const float4*)(x + idx);
    const float* ov = outv + b * D_ + col;
    float4 r = { xv.x + ov[0], xv.y + ov[1], xv.z + ov[2], xv.w + ov[3] };
    *(float4*)(y + idx) = r;
}

// ---------------------------------------------------------------------------
extern "C" void kernel_launch(void* const* d_in, const int* in_sizes, int n_in,
                              void* d_out, int out_size, void* d_ws, size_t ws_size,
                              hipStream_t stream) {
    const float* x     = (const float*)d_in[0];
    const float* ln_g  = (const float*)d_in[1];
    const float* ln_b  = (const float*)d_in[2];
    const float* lat   = (const float*)d_in[3];
    const float* w_lq  = (const float*)d_in[4];
    const float* b_lq  = (const float*)d_in[5];
    const float* w_k   = (const float*)d_in[6];
    const float* b_k   = (const float*)d_in[7];
    const float* w_v   = (const float*)d_in[8];
    const float* b_v   = (const float*)d_in[9];
    const float* w_lv  = (const float*)d_in[10];
    const float* b_lv  = (const float*)d_in[11];
    const float* w_out = (const float*)d_in[12];
    const float* b_out = (const float*)d_in[13];
    float* y = (float*)d_out;

    char* ws = (char*)d_ws;
    bf16*  h      = (bf16*) ws;  ws += (size_t)BS_ * D_ * sizeof(bf16);       // 128 MB
    bf16*  At     = (bf16*) ws;  ws += (size_t)N_  * D_ * sizeof(bf16);       // 1 MB
    float* scores = (float*)ws;  ws += (size_t)BS_ * N_ * sizeof(float);      // 32 MB
    float* q      = (float*)ws;  ws += (size_t)L_  * D_ * sizeof(float);
    float* cbias  = (float*)ws;  ws += (size_t)N_       * sizeof(float);
    float* pbar   = (float*)ws;  ws += (size_t)B_ * S_ * H_ * sizeof(float);  // 2 MB
    float* upart  = (float*)ws;  ws += (size_t)B_ * UCH_ * H_ * D_ * sizeof(float); // 32 MB
    float* u      = (float*)ws;  ws += (size_t)B_ * H_ * D_ * sizeof(float);
    float* pc     = (float*)ws;  ws += (size_t)B_ * D_ * sizeof(float);
    float* tv     = (float*)ws;  ws += (size_t)B_ * D_ * sizeof(float);
    float* ov     = (float*)ws;  ws += (size_t)B_ * D_ * sizeof(float);

    ln_kernel    <<<dim3(BS_),        256, 0, stream>>>(x, ln_g, ln_b, h);
    q_kernel     <<<dim3(8, L_),      256, 0, stream>>>(lat, w_lq, b_lq, q);
    afold_kernel <<<dim3(8, N_),      256, 0, stream>>>(q, w_k, At);
    cbias_kernel <<<dim3(1),          256, 0, stream>>>(q, b_k, cbias);
    scores_wmma_kernel<<<dim3(BS_/16), 128, 0, stream>>>(h, At, cbias, scores);
    softmax_kernel<<<dim3(B_, H_),    256, 0, stream>>>(scores, pbar);
    upart_kernel <<<dim3(B_, UCH_),   256, 0, stream>>>(h, pbar, upart);
    ured_kernel  <<<dim3(H_*D_/256, B_), 256, 0, stream>>>(upart, u);
    pc_kernel    <<<dim3(8, B_),      256, 0, stream>>>(u, w_v, b_v, pc);
    mv_kernel    <<<dim3(8, B_),      256, 0, stream>>>(pc, w_lv, b_lv, tv);
    mv_kernel    <<<dim3(8, B_),      256, 0, stream>>>(tv, w_out, b_out, ov);
    resid_kernel <<<dim3((size_t)B_*S_*D_/4/256), 256, 0, stream>>>(x, ov, y);
}
